// HQQLinearTorchWeightOnlyInt4_44890998178435
// MI455X (gfx1250) — matrix operand order repaired and checked by reading, not compiled, not run on hardware
//
#include <hip/hip_runtime.h>
#include <hip/hip_bf16.h>

typedef __bf16 bf16_t;
typedef __attribute__((ext_vector_type(4)))  __bf16 v4bf;
typedef __attribute__((ext_vector_type(8)))  __bf16 v8bf;
typedef __attribute__((ext_vector_type(16))) __bf16 v16bf;
typedef __attribute__((ext_vector_type(8)))  float  v8f;
typedef __attribute__((ext_vector_type(4)))  unsigned int u32x4;
typedef __attribute__((ext_vector_type(4)))  int i32x4;

// Problem constants (match reference)
constexpr int M_DIM = 4096;
constexpr int K_DIM = 4096;
constexpr int N_DIM = 11008;
constexpr int GROUP = 128;
constexpr int NG    = K_DIM / GROUP;   // 32 groups along K

// Tiling
constexpr int BM  = 128;               // M rows per block
constexpr int BN  = 64;                // N cols per block
constexpr int KB  = 128;               // K chunk == GROUP
constexpr int LDA = KB + 8;            // padded LDS row strides (break 64-bank alignment)
constexpr int LDB = KB + 8;
constexpr int LDC = BN + 8;            // epilogue staging stride
constexpr int KTILES = K_DIM / KB;     // 32

// ---- async global->LDS 16B copy (gfx1250 ASYNCcnt path), with safe fallback ----
__device__ __forceinline__ void async_copy_b128(bf16_t* lds_dst, const bf16_t* gsrc) {
#if __has_builtin(__builtin_amdgcn_global_load_async_to_lds_b128)
    __builtin_amdgcn_global_load_async_to_lds_b128(
        (__attribute__((address_space(1))) i32x4*)gsrc,
        (__attribute__((address_space(3))) i32x4*)lds_dst,
        0, 0);
#else
    *(u32x4*)lds_dst = *(const u32x4*)gsrc;
#endif
}

__device__ __forceinline__ void wait_async_lds() {
#if __has_builtin(__builtin_amdgcn_s_wait_asynccnt)
    __builtin_amdgcn_s_wait_asynccnt(0);
#elif __has_builtin(__builtin_amdgcn_global_load_async_to_lds_b128)
    asm volatile("s_wait_asynccnt 0x0" ::: "memory");
#endif
}

__global__ __launch_bounds__(256)
void hqq_int4_wmma_gemm(const bf16_t* __restrict__ x,
                        const int*    __restrict__ Wq,
                        const bf16_t* __restrict__ scales,
                        const bf16_t* __restrict__ zeros,
                        bf16_t*       __restrict__ out)
{
    // Double-buffered tiles: 2*(34816 + 17408) = 104448 B of the 320 KB WGP LDS
    __shared__ __align__(16) bf16_t sA[2][BM * LDA];  // x tiles
    __shared__ __align__(16) bf16_t sB[2][BN * LDB];  // dequantized W tiles

    const int tid  = threadIdx.x;
    const int lane = tid & 31;
    const int wid  = tid >> 5;
    const int wm   = wid & 3;          // 4 wave strips along M (32 rows each)
    const int wn   = wid >> 2;         // 2 wave strips along N (32 cols each)
    const int l15  = lane & 15;
    const int hs   = lane >> 4;        // lane half-select

    const int m0 = blockIdx.x * BM;    // M fastest: co-resident blocks share W rows via L2
    const int n0 = blockIdx.y * BN;

    const bf16_t* xbase = x  + (size_t)m0 * K_DIM;
    const int*    wbase = Wq + (size_t)n0 * K_DIM;

    v8f acc[2][2] = {};

    // ---- stage x tile (async DMA into LDS): 2048 chunks of 8 halves ----
    auto stage_x = [&](int kb, int buf) {
        const int k0 = kb * KB;
        #pragma unroll
        for (int i = 0; i < 8; ++i) {
            int c   = tid + i * 256;
            int r   = c >> 4;                 // 16 chunks per row
            int col = (c & 15) * 8;           // element offset
            async_copy_b128(&sA[buf][r * LDA + col],
                            xbase + (size_t)r * K_DIM + k0 + col);
        }
    };
    // ---- issue W_q global loads into registers (2048 chunks of 4 ints) ----
    auto load_w = [&](int kb, u32x4 (&q)[8]) {
        const int k0 = kb * KB;
        #pragma unroll
        for (int i = 0; i < 8; ++i) {
            int c = tid + i * 256;
            int r = c >> 5;                   // 32 chunks per row
            q[i] = *(const u32x4*)(wbase + (size_t)r * K_DIM + k0 + (c & 31) * 4);
        }
    };
    // ---- dequantize and store W tile to LDS ----
    auto dequant_store_w = [&](int kb, const u32x4 (&q)[8], int buf) {
        #pragma unroll
        for (int i = 0; i < 8; ++i) {
            int c   = tid + i * 256;
            int r   = c >> 5;
            int col = (c & 31) * 4;
            float s = (float)scales[(size_t)(n0 + r) * NG + kb];
            float z = (float)zeros [(size_t)(n0 + r) * NG + kb];
            v4bf w;
            w[0] = (__bf16)(((int)q[i][0] - 8) * s + z);
            w[1] = (__bf16)(((int)q[i][1] - 8) * s + z);
            w[2] = (__bf16)(((int)q[i][2] - 8) * s + z);
            w[3] = (__bf16)(((int)q[i][3] - 8) * s + z);
            *(v4bf*)&sB[buf][r * LDB + col] = w;
        }
    };

    // ---- prologue: fill buffer 0 ----
    {
        u32x4 q[8];
        load_w(0, q);
        stage_x(0, 0);
        dequant_store_w(0, q, 0);
        wait_async_lds();
        __syncthreads();
    }

    for (int kb = 0; kb < KTILES; ++kb) {
        const int cur = kb & 1;
        const int nxt = cur ^ 1;
        const bool have_next = (kb + 1 < KTILES);

        // issue next tile's global traffic first so it overlaps the WMMAs below
        u32x4 q[8];
        if (have_next) {
            load_w(kb + 1, q);
            stage_x(kb + 1, nxt);
        }

        // ---- compute on buffer `cur`: 4 k-steps of 32, 2x2 WMMA tiles per wave ----
        #pragma unroll
        for (int kk = 0; kk < KB; kk += 32) {
            v16bf af[2], bfm[2];
            // A frag (16x32 bf16): two 8-half runs at half*8 and 16+half*8
            #pragma unroll
            for (int i = 0; i < 2; ++i) {
                const bf16_t* pa = &sA[cur][(wm * 32 + i * 16 + l15) * LDA + kk + hs * 8];
                v8bf lo = *(const v8bf*)pa;
                v8bf hi = *(const v8bf*)(pa + 16);
                af[i] = __builtin_shufflevector(lo, hi,
                        0,1,2,3,4,5,6,7,8,9,10,11,12,13,14,15);
            }
            // B frag (32x16 bf16): two 8-half runs at half*8 and 16+half*8
            #pragma unroll
            for (int j = 0; j < 2; ++j) {
                const bf16_t* pb = &sB[cur][(wn * 32 + j * 16 + l15) * LDB + kk + hs * 8];
                v8bf lo = *(const v8bf*)pb;
                v8bf hi = *(const v8bf*)(pb + 16);
                bfm[j] = __builtin_shufflevector(lo, hi,
                        0,1,2,3,4,5,6,7,8,9,10,11,12,13,14,15);
            }
            #pragma unroll
            for (int i = 0; i < 2; ++i)
                #pragma unroll
                for (int j = 0; j < 2; ++j)
                    acc[i][j] = __builtin_amdgcn_wmma_f32_16x16x32_bf16(
                        false, af[i], false, bfm[j],
                        (short)0, acc[i][j], false, false);
        }

        // dequant next W tile after compute issue (global loads already in flight)
        if (have_next) dequant_store_w(kb + 1, q, nxt);

        wait_async_lds();
        __syncthreads();
    }

    // ---- epilogue: stage through LDS (overlay dead sA) for coalesced b128 stores ----
    bf16_t* stg = &sA[0][0];   // needs 128*72 halves = 18432 B <= 34816 B available
    #pragma unroll
    for (int i = 0; i < 2; ++i) {
        int rbase = wm * 32 + i * 16 + hs * 8;
        #pragma unroll
        for (int j = 0; j < 2; ++j) {
            int colb = wn * 32 + j * 16 + l15;
            #pragma unroll
            for (int v = 0; v < 8; ++v)
                stg[(rbase + v) * LDC + colb] = (__bf16)acc[i][j][v];
        }
    }
    __syncthreads();
    // 128 rows x 128B per row = 1024 16B chunks, 4 per thread, fully coalesced
    #pragma unroll
    for (int i = 0; i < 4; ++i) {
        int c   = tid + i * 256;
        int r   = c >> 3;
        int col = (c & 7) * 8;
        u32x4 v = *(const u32x4*)&stg[r * LDC + col];
        *(u32x4*)(out + (size_t)(m0 + r) * N_DIM + n0 + col) = v;
    }
}

extern "C" void kernel_launch(void* const* d_in, const int* in_sizes, int n_in,
                              void* d_out, int out_size, void* d_ws, size_t ws_size,
                              hipStream_t stream) {
    const bf16_t* x      = (const bf16_t*)d_in[0];
    const int*    Wq     = (const int*)   d_in[1];
    const bf16_t* scales = (const bf16_t*)d_in[2];
    const bf16_t* zeros  = (const bf16_t*)d_in[3];
    bf16_t*       out    = (bf16_t*)d_out;

    dim3 grid(M_DIM / BM, N_DIM / BN);   // 32 x 172
    hqq_int4_wmma_gemm<<<grid, 256, 0, stream>>>(x, Wq, scales, zeros, out);
}
